// DynamicConvolutionAttention_81423989997590
// MI455X (gfx1250) — matrix operand order repaired and checked by reading, not compiled, not run on hardware
//
#include <hip/hip_runtime.h>
#include <hip/hip_bf16.h>
#include <math.h>

typedef __attribute__((ext_vector_type(16))) _Float16 v16h;
typedef __attribute__((ext_vector_type(8)))  float    v8f;

union V16H { v16h v; _Float16 e[16]; };

#define B_DIM 128
#define T_DIM 4096
#define QD 1024
#define AD 128
#define DCK 168   // DC*DK = 8*21
#define PL 11
#define TILE_T 128

// Async global->LDS copy of one dword (CDNA5 GLOBAL_LOAD_ASYNC_TO_LDS_B32,
// tracked by ASYNCcnt). lds_off = byte offset of destination in LDS,
// gaddr = 64-bit global source address.
__device__ __forceinline__ void async_ld_lds_b32(unsigned lds_off, const float* src) {
  unsigned long long gaddr = (unsigned long long)(uintptr_t)src;
  asm volatile("global_load_async_to_lds_b32 %0, %1, off"
               :: "v"(lds_off), "v"(gaddr) : "memory");
}
__device__ __forceinline__ void wait_asynccnt0() {
  asm volatile("s_wait_asynccnt 0x0" ::: "memory");
}

// ---------------------------------------------------------------------------
// Kernel 1: H[128][128] = tanh(query @ Wq + bq)
// One wave per 16x16 output tile, K=1024 in steps of 32 via WMMA f16->f32.
// ---------------------------------------------------------------------------
__global__ __launch_bounds__(32) void k_qmlp(const float* __restrict__ query,
                                             const float* __restrict__ Wq,
                                             const float* __restrict__ bq,
                                             float* __restrict__ H) {
  const int N0 = blockIdx.x * 16, M0 = blockIdx.y * 16;
  const int lane = threadIdx.x;
  const int row = lane & 15, g = lane >> 4;
  v8f acc = {};
  for (int k0 = 0; k0 < QD; k0 += 32) {
    V16H A, Bm;
#pragma unroll
    for (int h = 0; h < 16; ++h) {
      // A (16x32 f16): half h of lane -> K = 16*(h>>3) + 8*g + (h&7)
      int Ka = k0 + 16 * (h >> 3) + 8 * g + (h & 7);
      A.e[h] = (_Float16)query[(size_t)(M0 + row) * QD + Ka];
      // B (32x16 f16): lane col = lane&15, K = 16*(h>>3) + 8*g + (h&7)
      int Kb = k0 + 16 * (h >> 3) + 8 * g + (h & 7);
      Bm.e[h] = (_Float16)Wq[(size_t)Kb * AD + (N0 + row)];
    }
    acc = __builtin_amdgcn_wmma_f32_16x16x32_f16(false, A.v, false, Bm.v,
                                                 (short)0, acc, false, false);
  }
  const int N = N0 + row;
  const float b = bq[N];
#pragma unroll
  for (int i = 0; i < 8; ++i) {
    int M = M0 + 8 * g + i;                 // C/D: vgpr i -> M = i (+8 for hi half)
    H[M * AD + N] = tanhf(acc[i] + b);
  }
}

// ---------------------------------------------------------------------------
// Kernel 2: G[128][168] = H @ Wk   (dynamic conv kernels per batch row)
// ---------------------------------------------------------------------------
__global__ __launch_bounds__(32) void k_dynk(const float* __restrict__ H,
                                             const float* __restrict__ Wk,
                                             float* __restrict__ G) {
  const int N0 = blockIdx.x * 16, M0 = blockIdx.y * 16;
  const int lane = threadIdx.x;
  const int row = lane & 15, g = lane >> 4;
  const int col = N0 + row;                 // up to 175; valid < 168
  v8f acc = {};
  for (int k0 = 0; k0 < AD; k0 += 32) {
    V16H A, Bm;
#pragma unroll
    for (int h = 0; h < 16; ++h) {
      int Ka = k0 + 16 * (h >> 3) + 8 * g + (h & 7);
      A.e[h] = (_Float16)H[(M0 + row) * AD + Ka];
      int Kb = k0 + 16 * (h >> 3) + 8 * g + (h & 7);
      Bm.e[h] = (_Float16)((col < DCK) ? Wk[Kb * DCK + col] : 0.0f);
    }
    acc = __builtin_amdgcn_wmma_f32_16x16x32_f16(false, A.v, false, Bm.v,
                                                 (short)0, acc, false, false);
  }
  if (col < DCK) {
#pragma unroll
    for (int i = 0; i < 8; ++i) G[(M0 + 8 * g + i) * DCK + col] = acc[i];
  }
}

// ---------------------------------------------------------------------------
// Kernel 3 (fused): static + dynamic 21-tap convs -> 16 features per (b,t),
// feat @ Wcat[16,128] + bdf via WMMA (K padded to 32), tanh, dot v,
// + log(clip(prior-conv)) -> energy, written to d_out.
// Block = 256 threads (8 waves), covers one b and 128 t positions.
// LDS staging uses CDNA5 async global->LDS copies (ASYNCcnt).
// ---------------------------------------------------------------------------
__global__ __launch_bounds__(256) void k_energy(
    const float* __restrict__ aw, const float* __restrict__ prior,
    const float* __restrict__ Wstatic, const float* __restrict__ Gmat,
    const float* __restrict__ Wsf, const float* __restrict__ Wdf,
    const float* __restrict__ bdf, const float* __restrict__ vvec,
    float* __restrict__ energy) {
  __shared__ float s_aw[TILE_T + 20];   // halo of 10 each side
  __shared__ float s_w[16 * 21];        // rows 0-7: static taps, 8-15: dynamic taps
  __shared__ float s_prior[PL];

  const int b = blockIdx.y;
  const int t0 = blockIdx.x * TILE_T;
  const int tid = threadIdx.x;

  // aw tile + halo: async copy for in-range elements, zero-fill the edges.
  if (tid < TILE_T + 20) {
    int gt = t0 - 10 + tid;
    if (gt >= 0 && gt < T_DIM) {
      unsigned off = (unsigned)(uintptr_t)&s_aw[tid];
      async_ld_lds_b32(off, aw + (size_t)b * T_DIM + gt);
    } else {
      s_aw[tid] = 0.0f;
    }
  }
  // combined tap table: [0..167] = Wstatic (8x21), [168..335] = G[b] (8x21)
  {
    int i = tid - 0;
    if (i >= 0 && i < 2 * DCK) {
      const float* src = (i < DCK) ? (Wstatic + i) : (Gmat + (size_t)b * DCK + (i - DCK));
      unsigned off = (unsigned)(uintptr_t)&s_w[i];
      async_ld_lds_b32(off, src);
    }
  }
  if (tid < PL) s_prior[tid] = prior[tid];

  // Prefetch next tile of this row into L2 (global_prefetch_b8).
  if (t0 + TILE_T < T_DIM && tid == 0)
    __builtin_prefetch(aw + (size_t)b * T_DIM + t0 + TILE_T, 0, 1);

  wait_asynccnt0();
  __syncthreads();

  const int lane = tid & 31;
  const int wave = tid >> 5;
  const int row = lane & 15, g = lane >> 4;
  const int tloc = wave * 16 + row;          // 0..127

  // A tile: 16 t-rows x 16 features (K 16..31 zero-padded).
  // half h<8 -> K = 8*g + h = feature index; h>=8 -> K>=16 -> 0.
  V16H A;
#pragma unroll
  for (int j = 0; j < 8; ++j) {
    const int f = 8 * g + j;
    const float* w = &s_w[f * 21];
    float acc = 0.0f;
#pragma unroll
    for (int k = 0; k < 21; ++k) acc += s_aw[tloc + k] * w[k];
    A.e[j] = (_Float16)acc;
    A.e[8 + j] = (_Float16)0.0f;
  }

  float p[8];
#pragma unroll
  for (int i = 0; i < 8; ++i) p[i] = 0.0f;
  const int colL = lane & 15;

#pragma unroll
  for (int n = 0; n < 8; ++n) {            // 8 N-tiles of 16 -> AD=128
    const int col = n * 16 + colL;
    V16H Bm;
#pragma unroll
    for (int h = 0; h < 16; ++h) {
      // B: lane col, K = 16*(h>>3) + 8*g + (h&7). h>=8 => K>=16 => zero pad.
      float w = (h < 8) ? ((g == 0) ? Wsf[h * AD + col] : Wdf[h * AD + col])
                         : 0.0f;
      Bm.e[h] = (_Float16)w;
    }
    const float bias = bdf[col];
    v8f c;
#pragma unroll
    for (int i = 0; i < 8; ++i) c[i] = bias;  // pre-load bias into accumulator
    c = __builtin_amdgcn_wmma_f32_16x16x32_f16(false, A.v, false, Bm.v,
                                               (short)0, c, false, false);
    const float vn = vvec[col];
#pragma unroll
    for (int i = 0; i < 8; ++i) p[i] += tanhf(c[i]) * vn;
  }

  // Reduce partial energies across the 16 lanes of each half-wave.
#pragma unroll
  for (int mask = 1; mask < 16; mask <<= 1) {
#pragma unroll
    for (int i = 0; i < 8; ++i) p[i] += __shfl_xor(p[i], mask, 32);
  }

  // Lanes 0..7 of each half write rows 8*g+0..7; fuse prior filter here.
  const int j = lane & 15;
  if (j < 8) {
    const int rowM = 8 * g + j;
    const int tl = wave * 16 + rowM;
    float pf = 0.0f;
#pragma unroll
    for (int k = 0; k < PL; ++k) pf += s_aw[tl + k] * s_prior[k];  // aw[t+k-10]*P[k]
    pf = logf(fmaxf(pf, 1e-6f));
    energy[(size_t)b * T_DIM + (t0 + tl)] = p[j] + pf;
  }
}

// ---------------------------------------------------------------------------
// Kernel 4: in-place softmax over each row of T
// ---------------------------------------------------------------------------
__global__ __launch_bounds__(256) void k_softmax(float* __restrict__ out) {
  __shared__ float red[256];
  const int b = blockIdx.x;
  float* row = out + (size_t)b * T_DIM;
  const int tid = threadIdx.x;

  float m = -INFINITY;
  for (int t = tid; t < T_DIM; t += 256) m = fmaxf(m, row[t]);
  red[tid] = m;
  __syncthreads();
  for (int s = 128; s > 0; s >>= 1) {
    if (tid < s) red[tid] = fmaxf(red[tid], red[tid + s]);
    __syncthreads();
  }
  m = red[0];
  __syncthreads();

  float sum = 0.0f;
  for (int t = tid; t < T_DIM; t += 256) sum += expf(row[t] - m);
  red[tid] = sum;
  __syncthreads();
  for (int s = 128; s > 0; s >>= 1) {
    if (tid < s) red[tid] += red[tid + s];
    __syncthreads();
  }
  const float inv = 1.0f / red[0];

  for (int t = tid; t < T_DIM; t += 256) row[t] = expf(row[t] - m) * inv;
}

// ---------------------------------------------------------------------------
extern "C" void kernel_launch(void* const* d_in, const int* in_sizes, int n_in,
                              void* d_out, int out_size, void* d_ws, size_t ws_size,
                              hipStream_t stream) {
  const float* query   = (const float*)d_in[0];   // [128,1024]
  const float* aw      = (const float*)d_in[1];   // [128,4096]
  const float* prior   = (const float*)d_in[2];   // [11]
  const float* Wq      = (const float*)d_in[3];   // [1024,128]
  const float* bq      = (const float*)d_in[4];   // [128]
  const float* Wk      = (const float*)d_in[5];   // [128,168]
  const float* Wstatic = (const float*)d_in[6];   // [8,1,21]
  const float* Wsf     = (const float*)d_in[7];   // [8,128]
  const float* Wdf     = (const float*)d_in[8];   // [8,128]
  const float* bdf     = (const float*)d_in[9];   // [128]
  const float* v       = (const float*)d_in[10];  // [128]
  float* out = (float*)d_out;                     // [128,4096]

  float* H = (float*)d_ws;                        // [128][128]
  float* G = H + B_DIM * AD;                      // [128][168]

  k_qmlp  <<<dim3(AD / 16, B_DIM / 16), 32, 0, stream>>>(query, Wq, bq, H);
  k_dynk  <<<dim3((DCK + 15) / 16, B_DIM / 16), 32, 0, stream>>>(H, Wk, G);
  k_energy<<<dim3(T_DIM / TILE_T, B_DIM), 256, 0, stream>>>(
      aw, prior, Wstatic, G, Wsf, Wdf, bdf, v, out);
  k_softmax<<<B_DIM, 256, 0, stream>>>(out);
}
